// MambaVisionLayer_Ours_15745350107171
// MI455X (gfx1250) — matrix operand order repaired and checked by reading, not compiled, not run on hardware
//
#include <hip/hip_runtime.h>
#include <hip/hip_bf16.h>

typedef __attribute__((ext_vector_type(16))) __bf16 v16bf;
typedef __attribute__((ext_vector_type(8)))  __bf16 v8bf;
typedef __attribute__((ext_vector_type(8)))  float  v8f;

#define D_MODEL   256
#define D_STATE   16
#define HEADDIM   16
#define D_INNER   512
#define NHEADS    32
#define CONV_DIM  544          // D_INNER + 2*D_STATE
#define D_IN_PROJ 1088         // 2*D_INNER + 2*D_STATE + NHEADS
#define BATCH     8
#define SEQLEN    4096
#define NTOK      (BATCH*SEQLEN)   // 32768
#define LCH       64               // scan chunk length
#define NCHUNK    (SEQLEN/LCH)     // 64

static __device__ __forceinline__ float silu(float v) {
    return v / (1.0f + __expf(-v));
}

// ---------------------------------------------------------------------------
// Weight pack: W[K][N] fp32 row-major  ->  Wt[N][K] bf16 (contiguous K per col)
// ---------------------------------------------------------------------------
__global__ void pack_w_kernel(const float* __restrict__ W, __bf16* __restrict__ Wt,
                              int K, int N) {
    int idx = blockIdx.x * 256 + threadIdx.x;
    if (idx >= K * N) return;
    int n = idx % N;
    int k = idx / N;
    Wt[(size_t)n * K + k] = (__bf16)W[idx];
}

// ---------------------------------------------------------------------------
// WMMA GEMM: C[M][ldc] = A[M][lda] * Wt^T   (Wt stored [N][K] bf16)
// One wave -> 16x64 tile (A fragment reused across 4 WMMA B tiles).
// Block = 8 waves -> 128 rows. grid.x = M/128, grid.y = N/64.
// ---------------------------------------------------------------------------
template <bool ABF16>
__global__ __launch_bounds__(256)
void wmma_gemm_kernel(const void* __restrict__ Aptr,
                      const __bf16* __restrict__ Wt,
                      float* __restrict__ C,
                      int lda, int ldc, int K) {
    const int lane = threadIdx.x & 31;
    const int wave = threadIdx.x >> 5;
    const int m0   = (blockIdx.x * 8 + wave) * 16;
    const int n0   = blockIdx.y * 64;
    const int krow = lane & 15;   // row (A) / col (B,C,D) within tile
    const int ksel = lane >> 4;   // which K-half this lane holds

    v8f acc0 = {}, acc1 = {}, acc2 = {}, acc3 = {};

    for (int k0 = 0; k0 < K; k0 += 32) {
        // ---- A fragment: 16x32 bf16 (contiguous K: k = k0 + 16*ksel + s) ----
        v16bf a;
        if (ABF16) {
            const __bf16* A  = (const __bf16*)Aptr;
            const __bf16* p0 = A + (size_t)(m0 + krow) * lda + k0 + ksel * 16;
            v8bf x0 = *(const v8bf*)p0;          // K = k0 + ksel*16 .. +7
            v8bf x1 = *(const v8bf*)(p0 + 8);    // K = k0+8+ksel*16 .. +7
#pragma unroll
            for (int i = 0; i < 8; ++i) { a[i] = x0[i]; a[8 + i] = x1[i]; }
        } else {
            const float* A  = (const float*)Aptr;
            const float* p0 = A + (size_t)(m0 + krow) * lda + k0 + ksel * 16;
            v8f f0 = *(const v8f*)p0;
            v8f f1 = *(const v8f*)(p0 + 8);
#pragma unroll
            for (int i = 0; i < 8; ++i) { a[i] = (__bf16)f0[i]; a[8 + i] = (__bf16)f1[i]; }
        }

        // ---- B fragments: 32x16 bf16, contiguous K from packed Wt[N][K] ----
        const __bf16* bb = Wt + (size_t)(n0 + krow) * K + k0 + ksel * 16;
        v16bf b0 = *(const v16bf*)(bb);
        v16bf b1 = *(const v16bf*)(bb + (size_t)16 * K);
        v16bf b2 = *(const v16bf*)(bb + (size_t)32 * K);
        v16bf b3 = *(const v16bf*)(bb + (size_t)48 * K);

        acc0 = __builtin_amdgcn_wmma_f32_16x16x32_bf16(false, a, false, b0, (short)0, acc0, false, false);
        acc1 = __builtin_amdgcn_wmma_f32_16x16x32_bf16(false, a, false, b1, (short)0, acc1, false, false);
        acc2 = __builtin_amdgcn_wmma_f32_16x16x32_bf16(false, a, false, b2, (short)0, acc2, false, false);
        acc3 = __builtin_amdgcn_wmma_f32_16x16x32_bf16(false, a, false, b3, (short)0, acc3, false, false);
    }

    // ---- store D: VGPR r -> row m0 + r + 8*ksel, col n0 + tile*16 + krow ----
    float* cp = C + (size_t)(m0 + ksel * 8) * ldc + n0 + krow;
#pragma unroll
    for (int r = 0; r < 8; ++r) {
        cp[(size_t)r * ldc +  0] = acc0[r];
        cp[(size_t)r * ldc + 16] = acc1[r];
        cp[(size_t)r * ldc + 32] = acc2[r];
        cp[(size_t)r * ldc + 48] = acc3[r];
    }
}

// ---------------------------------------------------------------------------
// Depthwise causal conv (k=4) + bias + SiLU over xBC slice of zxbcdt.
// ---------------------------------------------------------------------------
__global__ void conv_silu_kernel(const float* __restrict__ zx,
                                 const float* __restrict__ cw,
                                 const float* __restrict__ cb,
                                 float* __restrict__ xc) {
    int idx = blockIdx.x * 256 + threadIdx.x;   // over NTOK*CONV_DIM
    if (idx >= NTOK * CONV_DIM) return;
    int  c   = idx % CONV_DIM;
    long row = idx / CONV_DIM;                  // b*SEQLEN + t
    int  t   = (int)(row % SEQLEN);
    const float* col = zx + row * D_IN_PROJ + D_INNER + c;
    float acc = cb[c];
#pragma unroll
    for (int j = 0; j < 4; ++j) {
        int ts = t - 3 + j;
        if (ts >= 0) acc += cw[c * 4 + j] * col[(long)(j - 3) * D_IN_PROJ];
    }
    xc[row * (long)CONV_DIM + c] = silu(acc);
}

// ---------------------------------------------------------------------------
// dt = softplus(raw + dt_bias); dA = exp(dt * (-exp(A_log)))
// ---------------------------------------------------------------------------
__global__ void dt_kernel(const float* __restrict__ zx,
                          const float* __restrict__ dt_bias,
                          const float* __restrict__ A_log,
                          float* __restrict__ dtb,
                          float* __restrict__ dab) {
    int idx = blockIdx.x * 256 + threadIdx.x;   // over NTOK*NHEADS
    if (idx >= NTOK * NHEADS) return;
    int  h   = idx % NHEADS;
    long row = idx / NHEADS;
    float x  = zx[row * D_IN_PROJ + (D_IN_PROJ - NHEADS) + h] + dt_bias[h];
    float sp = (x > 20.0f) ? x : log1pf(__expf(x));
    float A  = -__expf(A_log[h]);
    dtb[idx] = sp;
    dab[idx] = __expf(sp * A);
}

// ---------------------------------------------------------------------------
// Chunked scan, pass A: per-(b,head,chunk) local scan with h0 = 0.
// Wave layout: lanes 0-15 hold p=lane with n=0..7; lanes 16-31 n=8..15.
// Writes: y_loc, per-t in-chunk decay cumprod s_t, chunk-final local state,
// and chunk decay product P.
// ---------------------------------------------------------------------------
__global__ __launch_bounds__(256)
void scan_chunk_kernel(const float* __restrict__ xc,
                       const float* __restrict__ dtb,
                       const float* __restrict__ dab,
                       float* __restrict__ yloc,
                       float* __restrict__ sbuf,
                       float* __restrict__ hend,
                       float* __restrict__ Pbuf) {
    const int b     = blockIdx.x;
    const int chunk = blockIdx.y;
    const int head  = blockIdx.z * 8 + (threadIdx.x >> 5);
    const int lane  = threadIdx.x & 31;
    const int p     = lane & 15;
    const int n0    = (lane >> 4) * 8;

    float h[8] = {0.f,0.f,0.f,0.f,0.f,0.f,0.f,0.f};
    float s = 1.0f;
    const int  t0      = chunk * LCH;
    const long rowbase = (long)b * SEQLEN;

    for (int t = t0; t < t0 + LCH; ++t) {
        const long r = rowbase + t;
        const float* xrow = xc + r * (long)CONV_DIM;
        if (t + 8 < t0 + LCH)                       // stream-prefetch next rows
            __builtin_prefetch(xrow + 8 * CONV_DIM + lane * 16, 0, 0);
        float xv  = xrow[head * HEADDIM + p];
        float4 B0 = *(const float4*)(xrow + D_INNER + n0);
        float4 B1 = *(const float4*)(xrow + D_INNER + n0 + 4);
        float4 C0 = *(const float4*)(xrow + D_INNER + D_STATE + n0);
        float4 C1 = *(const float4*)(xrow + D_INNER + D_STATE + n0 + 4);
        float dtv = dtb[r * NHEADS + head];
        float dav = dab[r * NHEADS + head];
        float cxb = dtv * xv;
        s *= dav;

        h[0] = dav * h[0] + cxb * B0.x;
        h[1] = dav * h[1] + cxb * B0.y;
        h[2] = dav * h[2] + cxb * B0.z;
        h[3] = dav * h[3] + cxb * B0.w;
        h[4] = dav * h[4] + cxb * B1.x;
        h[5] = dav * h[5] + cxb * B1.y;
        h[6] = dav * h[6] + cxb * B1.z;
        h[7] = dav * h[7] + cxb * B1.w;

        float acc = h[0]*C0.x + h[1]*C0.y + h[2]*C0.z + h[3]*C0.w
                  + h[4]*C1.x + h[5]*C1.y + h[6]*C1.z + h[7]*C1.w;
        acc += __shfl_xor(acc, 16, 32);
        if (lane < 16)
            yloc[r * (long)D_INNER + head * HEADDIM + p] = acc;
        if (lane == 0)
            sbuf[r * NHEADS + head] = s;
    }

    const long cb = (long)(b * NHEADS + head) * NCHUNK + chunk;
    float* hd = hend + cb * 256 + p * 16 + n0;
#pragma unroll
    for (int j = 0; j < 8; ++j) hd[j] = h[j];
    if (lane == 0) Pbuf[cb] = s;
}

// ---------------------------------------------------------------------------
// Chunked scan, pass B: inter-chunk state prefix (64 cheap serial steps).
// One wave per (b,head); writes entering state h_enter for every chunk.
// ---------------------------------------------------------------------------
__global__ __launch_bounds__(256)
void chunk_prefix_kernel(const float* __restrict__ hend,
                         const float* __restrict__ Pbuf,
                         float* __restrict__ hent) {
    const int bh   = blockIdx.x * 8 + (threadIdx.x >> 5);
    const int lane = threadIdx.x & 31;
    const int idx  = (lane & 15) * 16 + (lane >> 4) * 8;

    float h[8] = {0.f,0.f,0.f,0.f,0.f,0.f,0.f,0.f};
    for (int c = 0; c < NCHUNK; ++c) {
        const long cb = (long)bh * NCHUNK + c;
        float*       he = hent + cb * 256 + idx;
        const float* hd = hend + cb * 256 + idx;
        float P = Pbuf[cb];
#pragma unroll
        for (int j = 0; j < 8; ++j) {
            he[j] = h[j];                  // entering state for chunk c
            h[j]  = P * h[j] + hd[j];      // advance across chunk c
        }
    }
}

// ---------------------------------------------------------------------------
// Chunked scan, pass C: y_t = y_loc_t + s_t * (C_t . h_enter) + D * x_t.
// h_enter is chunk-constant -> held in registers; fully parallel.
// ---------------------------------------------------------------------------
__global__ __launch_bounds__(256)
void scan_fix_kernel(const float* __restrict__ xc,
                     const float* __restrict__ sbuf,
                     const float* __restrict__ hent,
                     const float* __restrict__ Dp,
                     float* __restrict__ y) {
    const int b     = blockIdx.x;
    const int chunk = blockIdx.y;
    const int head  = blockIdx.z * 8 + (threadIdx.x >> 5);
    const int lane  = threadIdx.x & 31;
    const int p     = lane & 15;
    const int n0    = (lane >> 4) * 8;
    const float Dv  = Dp[head];

    const long cb = (long)(b * NHEADS + head) * NCHUNK + chunk;
    const float* hep = hent + cb * 256 + p * 16 + n0;
    float he[8];
#pragma unroll
    for (int j = 0; j < 8; ++j) he[j] = hep[j];

    const int  t0      = chunk * LCH;
    const long rowbase = (long)b * SEQLEN;

    for (int t = t0; t < t0 + LCH; ++t) {
        const long r = rowbase + t;
        const float* xrow = xc + r * (long)CONV_DIM;
        if (t + 8 < t0 + LCH)
            __builtin_prefetch(xrow + 8 * CONV_DIM + lane * 16, 0, 0);
        float xv  = xrow[head * HEADDIM + p];
        float4 C0 = *(const float4*)(xrow + D_INNER + D_STATE + n0);
        float4 C1 = *(const float4*)(xrow + D_INNER + D_STATE + n0 + 4);
        float s   = sbuf[r * NHEADS + head];

        float dot = he[0]*C0.x + he[1]*C0.y + he[2]*C0.z + he[3]*C0.w
                  + he[4]*C1.x + he[5]*C1.y + he[6]*C1.z + he[7]*C1.w;
        dot += __shfl_xor(dot, 16, 32);
        if (lane < 16) {
            const long o = r * (long)D_INNER + head * HEADDIM + p;
            y[o] += s * dot + Dv * xv;
        }
    }
}

// ---------------------------------------------------------------------------
// t = y * silu(z); RMSNorm over D_INNER; write bf16 for out-proj WMMA GEMM.
// One wave per token row (512 elems -> 16 per lane), 8 rows per block.
// ---------------------------------------------------------------------------
__global__ __launch_bounds__(256)
void gate_norm_kernel(const float* __restrict__ y,
                      const float* __restrict__ zx,
                      const float* __restrict__ nw,
                      __bf16* __restrict__ tb) {
    const int  wave = threadIdx.x >> 5;
    const int  lane = threadIdx.x & 31;
    const long row  = (long)blockIdx.x * 8 + wave;
    const float* yr = y  + row * D_INNER;
    const float* zr = zx + row * D_IN_PROJ;   // z = first D_INNER cols

    float tv[16];
    float ss = 0.f;
#pragma unroll
    for (int i = 0; i < 16; ++i) {
        int c = lane + i * 32;
        float z = zr[c];
        float t = yr[c] * silu(z);
        tv[i] = t;
        ss += t * t;
    }
#pragma unroll
    for (int off = 16; off >= 1; off >>= 1) ss += __shfl_xor(ss, off, 32);
    float scale = rsqrtf(ss * (1.0f / D_INNER) + 1e-5f);

    __bf16* tr = tb + row * D_INNER;
#pragma unroll
    for (int i = 0; i < 16; ++i) {
        int c = lane + i * 32;
        tr[c] = (__bf16)(tv[i] * scale * nw[c]);
    }
}

// ---------------------------------------------------------------------------
extern "C" void kernel_launch(void* const* d_in, const int* in_sizes, int n_in,
                              void* d_out, int out_size, void* d_ws, size_t ws_size,
                              hipStream_t stream) {
    const float* u       = (const float*)d_in[0];
    const float* W_in    = (const float*)d_in[1];
    const float* conv_w  = (const float*)d_in[2];
    const float* conv_b  = (const float*)d_in[3];
    const float* dt_bias = (const float*)d_in[4];
    const float* A_log   = (const float*)d_in[5];
    const float* Dvec    = (const float*)d_in[6];
    const float* norm_w  = (const float*)d_in[7];
    const float* W_out   = (const float*)d_in[8];
    float* out = (float*)d_out;

    // Workspace carve-up (all chunks multiples of 256B). Total ~362 MB.
    char* ws = (char*)d_ws;
    float*  zxbcdt = (float*)ws;   ws += (size_t)NTOK * D_IN_PROJ * 4;            // 142.6 MB
    float*  xcv    = (float*)ws;   ws += (size_t)NTOK * CONV_DIM * 4;             //  71.3 MB
    float*  ybuf   = (float*)ws;   ws += (size_t)NTOK * D_INNER * 4;              //  67.1 MB
    float*  dtb    = (float*)ws;   ws += (size_t)NTOK * NHEADS * 4;               //   4.2 MB
    float*  dab    = (float*)ws;   ws += (size_t)NTOK * NHEADS * 4;               //   4.2 MB
    float*  sbuf   = (float*)ws;   ws += (size_t)NTOK * NHEADS * 4;               //   4.2 MB
    float*  hend   = (float*)ws;   ws += (size_t)BATCH*NHEADS*NCHUNK*256 * 4;     //  16.8 MB
    float*  hent   = (float*)ws;   ws += (size_t)BATCH*NHEADS*NCHUNK*256 * 4;     //  16.8 MB
    float*  Pbuf   = (float*)ws;   ws += (size_t)BATCH*NHEADS*NCHUNK * 4;         //  64 KB
    __bf16* tbf    = (__bf16*)ws;  ws += (size_t)NTOK * D_INNER * 2;              //  33.6 MB
    __bf16* wtin   = (__bf16*)ws;  ws += (size_t)D_IN_PROJ * D_MODEL * 2;
    __bf16* wtout  = (__bf16*)ws;  ws += (size_t)D_INNER * D_MODEL * 2;

    // 1) pack weights -> bf16, [N][K]
    pack_w_kernel<<<(D_MODEL * D_IN_PROJ + 255) / 256, 256, 0, stream>>>(W_in,  wtin,  D_MODEL, D_IN_PROJ);
    pack_w_kernel<<<(D_INNER * D_MODEL  + 255) / 256, 256, 0, stream>>>(W_out, wtout, D_INNER, D_MODEL);

    // 2) in-proj GEMM: zxbcdt[32768,1088] = u[32768,256] @ W_in
    wmma_gemm_kernel<false><<<dim3(NTOK / 128, D_IN_PROJ / 64), 256, 0, stream>>>(
        (const void*)u, wtin, zxbcdt, D_MODEL, D_IN_PROJ, D_MODEL);

    // 3) depthwise conv + SiLU ; dt/dA
    conv_silu_kernel<<<(NTOK * CONV_DIM + 255) / 256, 256, 0, stream>>>(zxbcdt, conv_w, conv_b, xcv);
    dt_kernel<<<(NTOK * NHEADS + 255) / 256, 256, 0, stream>>>(zxbcdt, dt_bias, A_log, dtb, dab);

    // 4) SSM scan, chunked: A (local scans) -> B (chunk prefix) -> C (fix-up)
    scan_chunk_kernel<<<dim3(BATCH, NCHUNK, NHEADS / 8), 256, 0, stream>>>(
        xcv, dtb, dab, ybuf, sbuf, hend, Pbuf);
    chunk_prefix_kernel<<<(BATCH * NHEADS) / 8, 256, 0, stream>>>(hend, Pbuf, hent);
    scan_fix_kernel<<<dim3(BATCH, NCHUNK, NHEADS / 8), 256, 0, stream>>>(
        xcv, sbuf, hent, Dvec, ybuf);

    // 5) gate + RMSNorm -> bf16
    gate_norm_kernel<<<NTOK / 8, 256, 0, stream>>>(ybuf, zxbcdt, norm_w, tbf);

    // 6) out-proj GEMM: out[32768,256] = t[32768,512] @ W_out
    wmma_gemm_kernel<true><<<dim3(NTOK / 128, D_MODEL / 64), 256, 0, stream>>>(
        (const void*)tbf, wtout, out, D_INNER, D_MODEL, D_INNER);
}
